// TimeMix_62843961475098
// MI455X (gfx1250) — compile-verified, matrix operand order checked by reading, not run on hardware
//
#include <hip/hip_runtime.h>
#include <hip/hip_bf16.h>

// ---------------------------------------------------------------------------
// RWKV TimeMix for MI455X (gfx1250, wave32, WMMA + TDM).
//   T=4096, C=2048.
//   GEMMs: v_wmma_f32_16x16x32_bf16, block tile 256x128, wave tile 64x64,
//          K-slab 64, LDS double-buffered via TENSOR_LOAD_TO_LDS (TDM).
//   Weights pre-transposed at bf16-convert time so A and B tiles are both
//   plain strided 2D tiles (TDM-friendly, no in-kernel transpose).
//   WKV recurrence: 3-phase chunked scan (128 chunks x 32 steps).
// ---------------------------------------------------------------------------

#define T_DIM 4096
#define C_DIM 2048
#define K_DIM 2048
#define N_DIM 2048
#define NCH   128
#define LCH   32

typedef __bf16 bf16;
typedef bf16  v16bf __attribute__((ext_vector_type(16)));
typedef bf16  v8bf  __attribute__((ext_vector_type(8)));
typedef float v8f   __attribute__((ext_vector_type(8)));
typedef unsigned int u32x4 __attribute__((ext_vector_type(4)));
typedef int i32x4 __attribute__((ext_vector_type(4)));
typedef int i32x8 __attribute__((ext_vector_type(8)));

#if defined(__HIP_DEVICE_COMPILE__)
# if __has_builtin(__builtin_amdgcn_tensor_load_to_lds) && __has_builtin(__builtin_amdgcn_s_wait_tensorcnt)
#  define HAVE_TDM 1
# endif
#endif
#ifndef HAVE_TDM
# define HAVE_TDM 0
#endif

// ------------------------ weight convert + transpose -----------------------
// Wt[n][k] = (bf16) W[k][n]   (2048 x 2048)
__global__ __launch_bounds__(256)
void transpose_bf16_kernel(const float* __restrict__ W, bf16* __restrict__ Wt) {
    __shared__ float tile[32][33];
    int bx = blockIdx.x * 32, by = blockIdx.y * 32;
    int tx = threadIdx.x & 31, ty = threadIdx.x >> 5;      // 32 x 8
#pragma unroll
    for (int j = 0; j < 32; j += 8)
        tile[ty + j][tx] = W[(size_t)(by + ty + j) * C_DIM + bx + tx];
    __syncthreads();
#pragma unroll
    for (int j = 0; j < 32; j += 8)
        Wt[(size_t)(bx + ty + j) * C_DIM + by + tx] = (bf16)tile[tx][ty + j];
}

// ------------------------------ time-shift mix -----------------------------
__global__ __launch_bounds__(256)
void mix_kernel(const float* __restrict__ x,
                const float* __restrict__ mk,
                const float* __restrict__ mv,
                const float* __restrict__ mr,
                bf16* __restrict__ xk,
                bf16* __restrict__ xv,
                bf16* __restrict__ xr) {
    int idx = blockIdx.x * 256 + threadIdx.x;              // over T*C
    int c = idx & (C_DIM - 1);
    int t = idx >> 11;
    float xc = x[idx];
    float xp = (t == 0) ? 0.0f : x[idx - C_DIM];
    float k = mk[c], v = mv[c], r = mr[c];
    xk[idx] = (bf16)(xc * k + xp * (1.0f - k));
    xv[idx] = (bf16)(xc * v + xp * (1.0f - v));
    xr[idx] = (bf16)(xc * r + xp * (1.0f - r));
}

// ------------------------------- WMMA GEMM ---------------------------------
// C[4096,2048] (f32) = A[4096,2048] (bf16 row-major) * Bt[2048,2048]^T
// Block: 256 thr = 8 waves (4 x 2). Block tile 256x128, wave tile 64x64,
// K-slab 64 -> 32 WMMAs / wave / barrier cycle. LDS rows padded to 72 bf16
// (144 B; bank step 36, gcd(36,64)=4 -> conflict-free ds_load_b128).
#define BTM 256
#define BTN 128
#define TKS 64
#define LSTR 72                     // padded LDS row stride (bf16 elements)

__device__ __forceinline__ v16bf load_frag(const bf16* base, int half) {
    union { v16bf v; v8bf h[2]; } u;
    u.h[0] = *(const v8bf*)(base + half * 8);
    u.h[1] = *(const v8bf*)(base + 16 + half * 8);
    return u.v;
}

#if HAVE_TDM
// Issue one TDM 2D tile load: tile_rows x 64 bf16 at gaddr (row stride 2048
// elements) into LDS at lds_addr, padding LDS rows 128B -> 144B.
__device__ __forceinline__ void tdm_load_tile(unsigned lds_addr,
                                              unsigned long long gaddr,
                                              unsigned tensor_rows,
                                              unsigned tile_rows) {
    u32x4 g0;
    g0[0] = 1u;                                            // count=1
    g0[1] = lds_addr;                                      // lds_addr
    g0[2] = (unsigned)gaddr;                               // global_addr lo
    g0[3] = (unsigned)((gaddr >> 32) & 0x1FFFFFFull) | (2u << 30);  // hi | type=2
    i32x8 g1;
    // data_size=2B(code1), pad_enable=1, pad_interval=32dw(code4), pad_amount=4dw(code3)
    g1[0] = (int)((1u << 16) | (1u << 20) | (4u << 22) | (3u << 25));
    g1[1] = (int)((unsigned)K_DIM << 16);                  // tensor_dim0 lo16
    g1[2] = (int)(tensor_rows << 16);                      // dim0 hi | tensor_dim1 lo16
    g1[3] = (int)((unsigned)TKS << 16);                    // dim1 hi | tile_dim0=64
    g1[4] = (int)tile_rows;                                // tile_dim1 | tile_dim2=0
    g1[5] = (int)K_DIM;                                    // tensor_dim0_stride lo
    g1[6] = 0;
    g1[7] = 0;
    i32x4 z4 = {0, 0, 0, 0};
#if __clang_major__ >= 23
    i32x8 z8 = {0, 0, 0, 0, 0, 0, 0, 0};
    __builtin_amdgcn_tensor_load_to_lds(g0, g1, z4, z4, z8, 0);
#else
    __builtin_amdgcn_tensor_load_to_lds(g0, g1, z4, z4, 0);
#endif
}
#endif

__global__ __launch_bounds__(256)
void gemm_bf16_kernel(const bf16* __restrict__ A,
                      const bf16* __restrict__ Bt,
                      float* __restrict__ Cout) {
    __shared__ bf16 lA[2][BTM * LSTR];       // 2 x 36 KB
    __shared__ bf16 lB[2][BTN * LSTR];       // 2 x 18 KB

    const int tid  = threadIdx.x;
    const int wid  = tid >> 5;
    const int lane = tid & 31;
    const int wr   = wid >> 1;               // 0..3 -> rows wr*64
    const int wc   = wid & 1;                // 0..1 -> cols wc*64
    const int m0   = blockIdx.y * BTM;
    const int n0   = blockIdx.x * BTN;
    const int l16  = lane & 15;
    const int half = lane >> 4;

    v8f acc[4][4] = {};

    const int KSTEPS = K_DIM / TKS;          // 32

#if HAVE_TDM
    const unsigned ldsA0 = (unsigned)(unsigned long long)(const void*)&lA[0][0];
    const unsigned ldsA1 = (unsigned)(unsigned long long)(const void*)&lA[1][0];
    const unsigned ldsB0 = (unsigned)(unsigned long long)(const void*)&lB[0][0];
    const unsigned ldsB1 = (unsigned)(unsigned long long)(const void*)&lB[1][0];
    const unsigned long long gA = (unsigned long long)A  + ((size_t)m0 * K_DIM) * 2ull;
    const unsigned long long gB = (unsigned long long)Bt + ((size_t)n0 * K_DIM) * 2ull;

    if (wid == 0) {                          // prologue: fill buffer 0
        tdm_load_tile(ldsA0, gA, T_DIM, BTM);
        tdm_load_tile(ldsB0, gB, N_DIM, BTN);
    }
    for (int ks = 0; ks < KSTEPS; ++ks) {
        const int buf = ks & 1;
        if (wid == 0) __builtin_amdgcn_s_wait_tensorcnt(0);
        __syncthreads();                     // buf ready; prev compute done
        if (wid == 0 && ks + 1 < KSTEPS) {   // prefetch next K-slab into other buf
            unsigned long long off = (unsigned long long)(ks + 1) * TKS * 2ull;
            tdm_load_tile(buf ? ldsA0 : ldsA1, gA + off, T_DIM, BTM);
            tdm_load_tile(buf ? ldsB0 : ldsB1, gB + off, N_DIM, BTN);
        }
#else
    for (int ks = 0; ks < KSTEPS; ++ks) {
        const int buf = 0;
        const int k0 = ks * TKS;
        __syncthreads();                     // protect previous compute
        {   // stage A: 256 rows x 64 bf16; one row (8 x b128) per thread
            const bf16* src = A + (size_t)(m0 + tid) * K_DIM + k0;
#pragma unroll
            for (int c = 0; c < 8; ++c)
                *(v8bf*)(&lA[0][tid * LSTR + c * 8]) = *(const v8bf*)(src + c * 8);
        }
        {   // stage B: 128 rows x 64 bf16; 4 x b128 per thread
            int row = tid & 127;
            int hseg = tid >> 7;             // 0/1
            const bf16* src = Bt + (size_t)(n0 + row) * K_DIM + k0 + hseg * 32;
#pragma unroll
            for (int c = 0; c < 4; ++c)
                *(v8bf*)(&lB[0][row * LSTR + hseg * 32 + c * 8]) =
                    *(const v8bf*)(src + c * 8);
        }
        __syncthreads();
#endif
        // ---- two K=32 groups per LDS slab; A-fragments software-pipelined ----
#pragma unroll
        for (int kh = 0; kh < 2; ++kh) {
            v16bf bfr[4];
#pragma unroll
            for (int j = 0; j < 4; ++j)
                bfr[j] = load_frag(
                    &lB[buf][(wc * 64 + j * 16 + l16) * LSTR + kh * 32], half);
            v16bf a_cur = load_frag(
                &lA[buf][(wr * 64 + l16) * LSTR + kh * 32], half);
#pragma unroll
            for (int i = 0; i < 4; ++i) {
                v16bf a_next = a_cur;
                if (i < 3)                   // prefetch next A frag under WMMAs
                    a_next = load_frag(
                        &lA[buf][(wr * 64 + (i + 1) * 16 + l16) * LSTR + kh * 32],
                        half);
#pragma unroll
                for (int j = 0; j < 4; ++j)
                    acc[i][j] = __builtin_amdgcn_wmma_f32_16x16x32_bf16(
                        false, a_cur, false, bfr[j], (short)0, acc[i][j],
                        false, false);
                a_cur = a_next;
            }
        }
    }

    // ---- store: one base pointer + compile-time immediate offsets ----
    float* outp = Cout + (size_t)(m0 + wr * 64 + 8 * half) * N_DIM
                       + (n0 + wc * 64 + l16);
#pragma unroll
    for (int i = 0; i < 4; ++i)
#pragma unroll
        for (int j = 0; j < 4; ++j)
#pragma unroll
            for (int r = 0; r < 8; ++r)
                outp[(size_t)(i * 16 + r) * N_DIM + j * 16] = acc[i][j][r];
}

// ------------------------------ WKV chunked scan ---------------------------
__global__ __launch_bounds__(256)
void wkv_phase1_kernel(const float* __restrict__ k, const float* __restrict__ v,
                       const float* __restrict__ time_decay,
                       float* __restrict__ Ploc, float* __restrict__ Qloc,
                       float* __restrict__ Oloc) {
    int c  = blockIdx.x * 256 + threadIdx.x;
    int ch = blockIdx.y;
    float w = -__expf(time_decay[c]);
    float p = 0.0f, q = 0.0f, o = -__builtin_inff();
    int t0 = ch * LCH;
    for (int t = 0; t < LCH; ++t) {
        size_t idx = (size_t)(t0 + t) * C_DIM + c;
        float kt = k[idx], vt = v[idx];
        float no = fmaxf(w + o, kt);
        float A  = __expf(w + o - no);
        float B  = __expf(kt - no);
        p = A * p + B * vt;
        q = A * q + B;
        o = no;
    }
    size_t s = (size_t)ch * C_DIM + c;
    Ploc[s] = p; Qloc[s] = q; Oloc[s] = o;
}

__global__ __launch_bounds__(256)
void wkv_phase2_kernel(const float* __restrict__ time_decay,
                       const float* __restrict__ Ploc, const float* __restrict__ Qloc,
                       const float* __restrict__ Oloc,
                       float* __restrict__ Pst, float* __restrict__ Qst,
                       float* __restrict__ Ost) {
    int c = blockIdx.x * 256 + threadIdx.x;
    float w  = -__expf(time_decay[c]);
    float wL = w * (float)LCH;
    float p = 0.0f, q = 0.0f, o = -__builtin_inff();
    for (int ch = 0; ch < NCH; ++ch) {
        size_t s = (size_t)ch * C_DIM + c;
        Pst[s] = p; Qst[s] = q; Ost[s] = o;
        float p2 = Ploc[s], q2 = Qloc[s], o2 = Oloc[s];
        float o1 = o + wL;
        float no = fmaxf(o1, o2);
        float A  = __expf(o1 - no);
        float B  = __expf(o2 - no);
        p = A * p + B * p2;
        q = A * q + B * q2;
        o = no;
    }
}

__global__ __launch_bounds__(256)
void wkv_phase3_kernel(const float* __restrict__ k, const float* __restrict__ v,
                       const float* __restrict__ r,
                       const float* __restrict__ time_decay,
                       const float* __restrict__ time_first,
                       const float* __restrict__ Pst, const float* __restrict__ Qst,
                       const float* __restrict__ Ost,
                       bf16* __restrict__ rwkv_bf) {
    int c  = blockIdx.x * 256 + threadIdx.x;
    int ch = blockIdx.y;
    float w = -__expf(time_decay[c]);
    float u = time_first[c];
    size_t s = (size_t)ch * C_DIM + c;
    float p = Pst[s], q = Qst[s], o = Ost[s];
    int t0 = ch * LCH;
    for (int t = 0; t < LCH; ++t) {
        size_t idx = (size_t)(t0 + t) * C_DIM + c;
        float kt = k[idx], vt = v[idx];
        float no = fmaxf(o, u + kt);
        float A  = __expf(o - no);
        float B  = __expf(u + kt - no);
        float y  = (A * p + B * vt) / (A * q + B);
        float sr = 1.0f / (1.0f + __expf(-r[idx]));
        rwkv_bf[idx] = (bf16)(sr * y);
        float no2 = fmaxf(w + o, kt);
        float A2  = __expf(w + o - no2);
        float B2  = __expf(kt - no2);
        p = A2 * p + B2 * vt;
        q = A2 * q + B2;
        o = no2;
    }
}

// ------------------------------- launcher ----------------------------------
extern "C" void kernel_launch(void* const* d_in, const int* in_sizes, int n_in,
                              void* d_out, int out_size, void* d_ws, size_t ws_size,
                              hipStream_t stream) {
    const float* x          = (const float*)d_in[0];
    const float* time_first = (const float*)d_in[1];
    const float* time_mix_k = (const float*)d_in[2];
    const float* time_mix_v = (const float*)d_in[3];
    const float* time_mix_r = (const float*)d_in[4];
    const float* time_decay = (const float*)d_in[5];
    const float* Wk         = (const float*)d_in[6];
    const float* Wv         = (const float*)d_in[7];
    const float* Wr         = (const float*)d_in[8];
    const float* Wo         = (const float*)d_in[9];
    float* out = (float*)d_out;

    const size_t TC = (size_t)T_DIM * C_DIM;
    const size_t CC = (size_t)C_DIM * C_DIM;
    const size_t NC = (size_t)NCH * C_DIM;

    char* p = (char*)d_ws;
    bf16* xk_bf = (bf16*)p; p += TC * sizeof(bf16);
    bf16* xv_bf = (bf16*)p; p += TC * sizeof(bf16);
    bf16* xr_bf = (bf16*)p; p += TC * sizeof(bf16);
    bf16* Wk_t  = (bf16*)p; p += CC * sizeof(bf16);   // transposed bf16 weights
    bf16* Wv_t  = (bf16*)p; p += CC * sizeof(bf16);
    bf16* Wr_t  = (bf16*)p; p += CC * sizeof(bf16);
    bf16* Wo_t  = (bf16*)p; p += CC * sizeof(bf16);
    float* kbuf = (float*)p; p += TC * sizeof(float);
    float* vbuf = (float*)p; p += TC * sizeof(float);
    float* rbuf = (float*)p; p += TC * sizeof(float);
    float* Ploc = (float*)p; p += NC * sizeof(float);
    float* Qloc = (float*)p; p += NC * sizeof(float);
    float* Oloc = (float*)p; p += NC * sizeof(float);
    float* Pst  = (float*)p; p += NC * sizeof(float);
    float* Qst  = (float*)p; p += NC * sizeof(float);
    float* Ost  = (float*)p; p += NC * sizeof(float);
    bf16* rwkv_bf = (bf16*)p; p += TC * sizeof(bf16);

    // 1. weights -> bf16, transposed (so GEMM B tiles are TDM-friendly)
    {
        dim3 tgrid(C_DIM / 32, C_DIM / 32);
        transpose_bf16_kernel<<<tgrid, 256, 0, stream>>>(Wk, Wk_t);
        transpose_bf16_kernel<<<tgrid, 256, 0, stream>>>(Wv, Wv_t);
        transpose_bf16_kernel<<<tgrid, 256, 0, stream>>>(Wr, Wr_t);
        transpose_bf16_kernel<<<tgrid, 256, 0, stream>>>(Wo, Wo_t);
    }

    // 2. time-shift mix -> bf16 activations
    mix_kernel<<<(int)(TC / 256), 256, 0, stream>>>(
        x, time_mix_k, time_mix_v, time_mix_r, xk_bf, xv_bf, xr_bf);

    // 3. k / v / r GEMMs
    dim3 ggrid(N_DIM / BTN, T_DIM / BTM);     // (16, 16)
    gemm_bf16_kernel<<<ggrid, 256, 0, stream>>>(xk_bf, Wk_t, kbuf);
    gemm_bf16_kernel<<<ggrid, 256, 0, stream>>>(xv_bf, Wv_t, vbuf);
    gemm_bf16_kernel<<<ggrid, 256, 0, stream>>>(xr_bf, Wr_t, rbuf);

    // 4. chunked WKV scan
    dim3 sgrid(C_DIM / 256, NCH);
    wkv_phase1_kernel<<<sgrid, 256, 0, stream>>>(kbuf, vbuf, time_decay, Ploc, Qloc, Oloc);
    wkv_phase2_kernel<<<C_DIM / 256, 256, 0, stream>>>(time_decay, Ploc, Qloc, Oloc,
                                                       Pst, Qst, Ost);
    wkv_phase3_kernel<<<sgrid, 256, 0, stream>>>(kbuf, vbuf, rbuf, time_decay, time_first,
                                                 Pst, Qst, Ost, rwkv_bf);

    // 5. output GEMM: rwkv @ Wo -> out (fp32)
    gemm_bf16_kernel<<<ggrid, 256, 0, stream>>>(rwkv_bf, Wo_t, out);
}
